// vTransformer_2181843386554
// MI455X (gfx1250) — matrix operand order repaired and checked by reading, not compiled, not run on hardware
//
#include <hip/hip_runtime.h>
#include <stdint.h>

// ---------------------------------------------------------------------------
// Problem constants (match reference)
// ---------------------------------------------------------------------------
#define B_    2
#define N_    4096
#define INCH  128
#define DIM   256
#define KNN   16
#define POSH  64
#define HID   1024
#define EPS_  1e-5f

typedef __attribute__((ext_vector_type(16))) __bf16 v16bf;
typedef __attribute__((ext_vector_type(8)))  float  v8f;

// ---------------------------------------------------------------------------
// bf16 helpers (bit-exact RNE, no reliance on native __bf16 casts)
// ---------------------------------------------------------------------------
__device__ inline __bf16 f2bf(float f) {
  union { float f; uint32_t u; } a; a.f = f;
  uint32_t r = a.u + 0x7FFFu + ((a.u >> 16) & 1u);
  uint16_t h = (uint16_t)(r >> 16);
  __bf16 out; __builtin_memcpy(&out, &h, 2); return out;
}
__device__ inline float bf2f(__bf16 b) {
  uint16_t h; __builtin_memcpy(&h, &b, 2);
  union { uint32_t u; float f; } a; a.u = ((uint32_t)h) << 16; return a.f;
}

// ---------------------------------------------------------------------------
// WMMA fragment loaders (16x16x32 bf16, wave32; contiguous k order used for both A and B)
//   A (16xK=32, 16-bit): lane L -> row = L&15, kg = L>>4;
//     halfs 0..15 = K[k0 + kg*16 .. +16)
//   B (K=32 x 16, 16-bit): lane L -> col = L&15, kg = L>>4;
//     halfs 0..15 = K[k0 + kg*16 .. +16)
//   C/D f32: lane L -> col = L&15; VGPR j -> row = j + 8*(L>>4)
// ---------------------------------------------------------------------------
union FragU { uint4 u4[2]; v16bf v; };

__device__ inline v16bf load_frag_a(const __bf16* __restrict__ W, int ldw,
                                    int rowBase, int k0, int lane) {
  int m  = rowBase + (lane & 15);
  int kg = lane >> 4;
  const __bf16* p = W + (size_t)m * ldw + k0 + kg * 16;
  FragU f;
  f.u4[0] = *(const uint4*)(p);
  f.u4[1] = *(const uint4*)(p + 8);
  return f.v;
}

__device__ inline v16bf load_frag_b_rm(const __bf16* __restrict__ X, int ldx,
                                       int colBase, int k0, int lane) {
  int gc = colBase + (lane & 15);
  int kg = lane >> 4;
  const __bf16* p = X + (size_t)gc * ldx + k0 + kg * 16;
  FragU f;
  f.u4[0] = *(const uint4*)(p);
  f.u4[1] = *(const uint4*)(p + 8);
  return f.v;
}

__device__ inline v16bf load_frag_b_lds(const __bf16* S, int stride,
                                        int colBase, int k0, int lane) {
  int c  = colBase + (lane & 15);
  int kg = lane >> 4;
  const __bf16* p = S + c * stride + k0 + kg * 16;
  FragU f;
  f.u4[0] = *(const uint4*)(p);
  f.u4[1] = *(const uint4*)(p + 8);
  return f.v;
}

__device__ inline v8f wmma_bf16(v16bf a, v16bf b, v8f c) {
  return __builtin_amdgcn_wmma_f32_16x16x32_bf16(
      /*neg_a=*/false, a, /*neg_b=*/false, b,
      /*c_mod=*/(short)0, c, /*reuse_a=*/false, /*reuse_b=*/false);
}

// ---------------------------------------------------------------------------
// CDNA5 async global->LDS copy (ASYNCcnt-tracked, no VGPR round trip).
// LDS destination address = low 32 bits of the generic shared pointer
// (ISA 10.2: LDS aperture addresses truncate to the in-wave LDS offset).
// ---------------------------------------------------------------------------
__device__ inline void async_load_lds_b32(void* lds_dst, const void* gsrc) {
  uint32_t l = (uint32_t)(uintptr_t)lds_dst;
  uint64_t g = (uint64_t)(uintptr_t)gsrc;
  asm volatile("global_load_async_to_lds_b32 %0, %1, off"
               :: "v"(l), "v"(g) : "memory");
}
__device__ inline void wait_asynccnt0() {
  asm volatile("s_wait_asynccnt 0" ::: "memory");
}

// ---------------------------------------------------------------------------
// Small conversion / precompute kernels
// ---------------------------------------------------------------------------
__global__ void cvt_bf16_kernel(const float* __restrict__ s, __bf16* __restrict__ d, int n) {
  int i = blockIdx.x * blockDim.x + threadIdx.x;
  if (i < n) d[i] = f2bf(s[i]);
}

// x (B,C=128,N) f32  ->  xb (B,N,128) bf16 (point-major)
__global__ void transpose_x_kernel(const float* __restrict__ x, __bf16* __restrict__ xb) {
  int i = blockIdx.x * blockDim.x + threadIdx.x;      // over B*N*128, c fastest
  int c = i & (INCH - 1);
  int n = (i >> 7) & (N_ - 1);
  int b = i >> 19;
  xb[i] = f2bf(x[((size_t)b * INCH + c) * N_ + n]);
}

// Fold conv biases into eval-mode BN scale/shift.
__global__ void precompute_bn_kernel(
    const float* gp, const float* btp, const float* mp, const float* vp, const float* bp1,
    const float* ga, const float* bta, const float* ma, const float* va, const float* ba1,
    float* pscale, float* pshift, float* ascale, float* ashift) {
  int i = blockIdx.x * blockDim.x + threadIdx.x;
  if (i < POSH) {
    float s = gp[i] * rsqrtf(vp[i] + EPS_);
    pscale[i] = s;
    pshift[i] = btp[i] + (bp1[i] - mp[i]) * s;
  }
  if (i < HID) {
    float s = ga[i] * rsqrtf(va[i] + EPS_);
    ascale[i] = s;
    ashift[i] = bta[i] + (ba1[i] - ma[i]) * s;
  }
}

// ---------------------------------------------------------------------------
// Generic WMMA bf16 GEMM:  Y(D x BN) = W(D x C) * X(BN x C)^T + bias
// Block = 8 waves covering 8 consecutive row-tiles of ONE col-tile, so all
// waves consume identical B (activation) fragments -> 8x L0 reuse on the
// large activation stream; weights stay L2-resident.
// Optional outputs: f32 point-major, bf16 point-major, f32 channel-major
// (with fused residual add) for the final layer.
// ---------------------------------------------------------------------------
__global__ __launch_bounds__(256) void gemm_bf16_wmma(
    const __bf16* __restrict__ W, const __bf16* __restrict__ X,
    const float* __restrict__ bias,
    int D, int C, int BN, int Npts,
    float* __restrict__ outF32pm, __bf16* __restrict__ outBf16pm,
    float* __restrict__ outF32cm, const float* __restrict__ residual_cm) {
  int lane = threadIdx.x & 31;
  int w    = threadIdx.x >> 5;               // 0..7
  int colTiles = BN >> 4;
  int rg      = blockIdx.x / colTiles;       // row-group of 8 row-tiles
  int colTile = blockIdx.x - rg * colTiles;
  int rowTile = rg * 8 + w;
  if (rowTile >= (D >> 4)) return;           // uniform per wave; EXEC all-ones
  int rowBase = rowTile << 4, colBase = colTile << 4;

  v8f acc = {0, 0, 0, 0, 0, 0, 0, 0};
  for (int k0 = 0; k0 < C; k0 += 32) {
    v16bf a = load_frag_a(W, C, rowBase, k0, lane);
    v16bf b = load_frag_b_rm(X, C, colBase, k0, lane);
    acc = wmma_bf16(a, b, acc);
  }

  int kg  = lane >> 4;
  int col = colBase + (lane & 15);
  int bb  = col / Npts;
  int nn  = col - bb * Npts;
#pragma unroll
  for (int j = 0; j < 8; ++j) {
    int d = rowBase + j + 8 * kg;
    float v = acc[j] + bias[d];
    if (outF32pm)  outF32pm[(size_t)col * D + d]  = v;
    if (outBf16pm) outBf16pm[(size_t)col * D + d] = f2bf(v);
    if (outF32cm) {
      size_t o = ((size_t)bb * D + d) * Npts + nn;
      outF32cm[o] = v + (residual_cm ? residual_cm[o] : 0.0f);
    }
  }
}

// ---------------------------------------------------------------------------
// Brute-force KNN (K=16, includes self). One thread per query point.
// Candidate tiles staged through LDS with CDNA5 async global->LDS copies.
// ---------------------------------------------------------------------------
__global__ __launch_bounds__(256) void knn_kernel(const float* __restrict__ pos,
                                                  int* __restrict__ nidx) {
  __shared__ float sx[256], sy[256], sz[256];
  int tid = threadIdx.x;
  int b   = blockIdx.x / (N_ / 256);
  int n   = (blockIdx.x % (N_ / 256)) * 256 + tid;
  const float* pb = pos + (size_t)b * 3 * N_;
  float qx = pb[n], qy = pb[N_ + n], qz = pb[2 * N_ + n];

  float bd[KNN]; int bi[KNN];
#pragma unroll
  for (int j = 0; j < KNN; ++j) { bd[j] = 3.4e38f; bi[j] = 0; }

  for (int m0 = 0; m0 < N_; m0 += 256) {
    __syncthreads();                      // previous tile fully consumed
    async_load_lds_b32(&sx[tid], pb + m0 + tid);
    async_load_lds_b32(&sy[tid], pb + N_ + m0 + tid);
    async_load_lds_b32(&sz[tid], pb + 2 * N_ + m0 + tid);
    wait_asynccnt0();                     // this wave's LDS writes landed
    __syncthreads();                      // visible across the workgroup
    for (int j = 0; j < 256; ++j) {
      float dx = qx - sx[j], dy = qy - sy[j], dz = qz - sz[j];
      float d2 = dx * dx + dy * dy + dz * dz;
      if (d2 < bd[KNN - 1]) {
        bd[KNN - 1] = d2; bi[KNN - 1] = m0 + j;
#pragma unroll
        for (int t = KNN - 1; t > 0; --t) {
          if (bd[t] < bd[t - 1]) {
            float td = bd[t]; bd[t] = bd[t - 1]; bd[t - 1] = td;
            int   ti = bi[t]; bi[t] = bi[t - 1]; bi[t - 1] = ti;
          }
        }
      }
    }
  }
#pragma unroll
  for (int j = 0; j < KNN; ++j) nidx[((size_t)b * N_ + n) * KNN + j] = bi[j];
}

// ---------------------------------------------------------------------------
// Fused grouped attention. One workgroup (256 thr = 8 waves) handles PTS=2
// points => 32 columns (point*16 + neighbor).
//  Phase 1: gather + pos-MLP (VALU) -> T = bf16(qk_rel + pe), VAL = bf16(v+pe)
//           in LDS (col-major over channels, padded stride).
//  Phase 2: attn MLP: H = relu(bn(wa1 * T)) chunked 64 rows at a time through
//           LDS; logits = wa2 * H accumulated in WMMA f32 tiles.
//  Phase 3: softmax over the 16 neighbors (16-lane shfl_xor butterflies; the
//           16x16 C-tile puts one point's K=16 columns in lanes 0-15 / 16-31)
//           fused with the attn*value reduction -> agg (B,N,256) bf16.
// ---------------------------------------------------------------------------
#define TSTR 280   // 256 + 24 pad halfs: 16B-aligned rows, conflict-free spread
#define HSTR 72    // 64 + 8 pad halfs

__global__ __launch_bounds__(256) void fused_attn_kernel(
    const float* __restrict__ pos, const int* __restrict__ nidx,
    const float* __restrict__ qf, const float* __restrict__ kf, const float* __restrict__ vf,
    const float* __restrict__ wp1, const float* __restrict__ wp2, const float* __restrict__ bp2,
    const float* __restrict__ pscale, const float* __restrict__ pshift,
    const __bf16* __restrict__ wa1b, const __bf16* __restrict__ wa2b,
    const float* __restrict__ ascale, const float* __restrict__ ashift,
    const float* __restrict__ ba2,
    __bf16* __restrict__ aggB) {
  __shared__ __align__(16) __bf16 T[32 * TSTR];
  __shared__ __align__(16) __bf16 VAL[32 * TSTR];
  __shared__ __align__(16) __bf16 HS[32 * HSTR];

  int tid = threadIdx.x;
  int b   = blockIdx.x / (N_ / 2);
  int n0  = (blockIdx.x % (N_ / 2)) * 2;

  // ---- Phase 1: gather + pos MLP ------------------------------------------
  {
    int col = tid & 31;          // point_local*16 + k
    int seg = tid >> 5;          // 8 segments of 32 channels each
    int p   = col >> 4, kk = col & 15;
    int n   = n0 + p;
    int m   = nidx[((size_t)b * N_ + n) * KNN + kk];
    const float* posb = pos + (size_t)b * 3 * N_;
    float pr0 = posb[n] - posb[m];
    float pr1 = posb[N_ + n] - posb[N_ + m];
    float pr2 = posb[2 * N_ + n] - posb[2 * N_ + m];

    float h1[POSH];
#pragma unroll
    for (int d = 0; d < POSH; ++d) {
      float z = wp1[d * 3] * pr0 + wp1[d * 3 + 1] * pr1 + wp1[d * 3 + 2] * pr2;
      z = z * pscale[d] + pshift[d];
      h1[d] = z > 0.0f ? z : 0.0f;
    }
    const float* qrow = qf + ((size_t)b * N_ + n) * DIM;
    const float* krow = kf + ((size_t)b * N_ + m) * DIM;
    const float* vrow = vf + ((size_t)b * N_ + m) * DIM;
    for (int c = seg * 32; c < seg * 32 + 32; ++c) {
      float pe = bp2[c];
      const float* w2 = wp2 + c * POSH;
#pragma unroll
      for (int d = 0; d < POSH; ++d) pe += w2[d] * h1[d];
      T[col * TSTR + c]   = f2bf(qrow[c] - krow[c] + pe);
      VAL[col * TSTR + c] = f2bf(vrow[c] + pe);
    }
  }
  __syncthreads();

  // ---- Phase 2: attn MLP via WMMA -----------------------------------------
  int lane = tid & 31, w = tid >> 5;
  v8f acc2[4];
#pragma unroll
  for (int i = 0; i < 4; ++i) acc2[i] = (v8f){0, 0, 0, 0, 0, 0, 0, 0};

  for (int chunk = 0; chunk < HID / 64; ++chunk) {
    // Layer 1: H_chunk(64x32) = relu(bn(wa1[chunk] * T)); one tile per wave.
    {
      int rt = w >> 1, ct = w & 1;
      v8f hacc = {0, 0, 0, 0, 0, 0, 0, 0};
      for (int k0 = 0; k0 < DIM; k0 += 32) {
        v16bf A  = load_frag_a(wa1b, DIM, chunk * 64 + rt * 16, k0, lane);
        v16bf Bm = load_frag_b_lds(T, TSTR, ct * 16, k0, lane);
        hacc = wmma_bf16(A, Bm, hacc);
      }
      if (chunk + 1 < HID / 64)  // stream next chunk's weights toward L1/L2
        __builtin_prefetch(wa1b + (size_t)((chunk + 1) * 64 + (lane & 15)) * DIM, 0, 1);
      int kgl = lane >> 4;
      int cc  = ct * 16 + (lane & 15);
#pragma unroll
      for (int j = 0; j < 8; ++j) {
        int lr   = rt * 16 + j + 8 * kgl;
        int grow = chunk * 64 + lr;
        float z  = hacc[j] * ascale[grow] + ashift[grow];
        HS[cc * HSTR + lr] = f2bf(z > 0.0f ? z : 0.0f);
      }
    }
    __syncthreads();
    // Layer 2: logits(256x32) += wa2[:, chunk] * H_chunk; 4 tiles per wave.
#pragma unroll
    for (int i = 0; i < 4; ++i) {
      int rt2 = w * 2 + (i >> 1), ct2 = i & 1;
#pragma unroll
      for (int kk = 0; kk < 2; ++kk) {
        v16bf A  = load_frag_a(wa2b, HID, rt2 * 16, chunk * 64 + kk * 32, lane);
        v16bf Bm = load_frag_b_lds(HS, HSTR, ct2 * 16, kk * 32, lane);
        acc2[i] = wmma_bf16(A, Bm, acc2[i]);
      }
    }
    __syncthreads();
  }

  // ---- Phase 3: softmax over K=16 + weighted aggregation ------------------
  int kg = lane >> 4;
#pragma unroll
  for (int i = 0; i < 4; ++i) {
    int rt2 = w * 2 + (i >> 1), ct2 = i & 1;
    int col = ct2 * 16 + (lane & 15);
    int n   = n0 + ct2;
#pragma unroll
    for (int j = 0; j < 8; ++j) {
      int d = rt2 * 16 + j + 8 * kg;
      float logit = acc2[i][j] + ba2[d];
      float mx = logit;
      mx = fmaxf(mx, __shfl_xor(mx, 1, 32));
      mx = fmaxf(mx, __shfl_xor(mx, 2, 32));
      mx = fmaxf(mx, __shfl_xor(mx, 4, 32));
      mx = fmaxf(mx, __shfl_xor(mx, 8, 32));
      float e   = __expf(logit - mx);
      float val = bf2f(VAL[col * TSTR + d]);
      float s = e, sv = e * val;
      s += __shfl_xor(s, 1, 32);  sv += __shfl_xor(sv, 1, 32);
      s += __shfl_xor(s, 2, 32);  sv += __shfl_xor(sv, 2, 32);
      s += __shfl_xor(s, 4, 32);  sv += __shfl_xor(sv, 4, 32);
      s += __shfl_xor(s, 8, 32);  sv += __shfl_xor(sv, 8, 32);
      if ((lane & 15) == 0) {
        aggB[((size_t)b * N_ + n) * DIM + d] = f2bf(sv / s);
      }
    }
  }
}

// ---------------------------------------------------------------------------
// Host launcher
// ---------------------------------------------------------------------------
extern "C" void kernel_launch(void* const* d_in, const int* in_sizes, int n_in,
                              void* d_out, int out_size, void* d_ws, size_t ws_size,
                              hipStream_t stream) {
  const float* x       = (const float*)d_in[0];
  const float* pos     = (const float*)d_in[1];
  const float* w_start = (const float*)d_in[2];
  const float* b_start = (const float*)d_in[3];
  const float* w_key   = (const float*)d_in[4];
  const float* b_key   = (const float*)d_in[5];
  const float* w_query = (const float*)d_in[6];
  const float* b_query = (const float*)d_in[7];
  const float* w_value = (const float*)d_in[8];
  const float* b_value = (const float*)d_in[9];
  const float* wp1     = (const float*)d_in[10];
  const float* bp1     = (const float*)d_in[11];
  const float* gp      = (const float*)d_in[12];
  const float* btp     = (const float*)d_in[13];
  const float* mp      = (const float*)d_in[14];
  const float* vp      = (const float*)d_in[15];
  const float* wp2     = (const float*)d_in[16];
  const float* bp2     = (const float*)d_in[17];
  const float* wa1     = (const float*)d_in[18];
  const float* ba1     = (const float*)d_in[19];
  const float* ga      = (const float*)d_in[20];
  const float* bta     = (const float*)d_in[21];
  const float* ma      = (const float*)d_in[22];
  const float* va      = (const float*)d_in[23];
  const float* wa2     = (const float*)d_in[24];
  const float* ba2     = (const float*)d_in[25];
  const float* w_end   = (const float*)d_in[26];
  const float* b_end   = (const float*)d_in[27];
  float* out = (float*)d_out;

  // Workspace carve-up
  char* wsb = (char*)d_ws;
  size_t off = 0;
  auto alloc = [&](size_t bytes) -> void* {
    void* p = wsb + off;
    off = (off + bytes + 255) & ~(size_t)255;
    return p;
  };
  const size_t BN = (size_t)B_ * N_;
  __bf16* xb      = (__bf16*)alloc(BN * INCH * 2);   // x, point-major bf16
  __bf16* xs_b    = (__bf16*)alloc(BN * DIM * 2);    // start output, bf16 pm
  float*  qf      = (float*)alloc(BN * DIM * 4);
  float*  kfv     = (float*)alloc(BN * DIM * 4);
  float*  vfv     = (float*)alloc(BN * DIM * 4);
  int*    nidx    = (int*)alloc(BN * KNN * 4);
  __bf16* aggB    = (__bf16*)alloc(BN * DIM * 2);
  __bf16* w_start_b = (__bf16*)alloc((size_t)DIM * INCH * 2);
  __bf16* w_key_b   = (__bf16*)alloc((size_t)DIM * DIM * 2);
  __bf16* w_query_b = (__bf16*)alloc((size_t)DIM * DIM * 2);
  __bf16* w_value_b = (__bf16*)alloc((size_t)DIM * DIM * 2);
  __bf16* wa1_b     = (__bf16*)alloc((size_t)HID * DIM * 2);
  __bf16* wa2_b     = (__bf16*)alloc((size_t)DIM * HID * 2);
  __bf16* w_end_b   = (__bf16*)alloc((size_t)INCH * DIM * 2);
  float* pscale = (float*)alloc(POSH * 4);
  float* pshift = (float*)alloc(POSH * 4);
  float* ascale = (float*)alloc(HID * 4);
  float* ashift = (float*)alloc(HID * 4);

  auto cvt = [&](const float* s, __bf16* d, int n) {
    cvt_bf16_kernel<<<(n + 255) / 256, 256, 0, stream>>>(s, d, n);
  };

  // 1) Precision staging
  cvt(w_start, w_start_b, DIM * INCH);
  cvt(w_key,   w_key_b,   DIM * DIM);
  cvt(w_query, w_query_b, DIM * DIM);
  cvt(w_value, w_value_b, DIM * DIM);
  cvt(wa1,     wa1_b,     HID * DIM);
  cvt(wa2,     wa2_b,     DIM * HID);
  cvt(w_end,   w_end_b,   INCH * DIM);
  transpose_x_kernel<<<(int)(BN * INCH / 256), 256, 0, stream>>>(x, xb);
  precompute_bn_kernel<<<4, 256, 0, stream>>>(gp, btp, mp, vp, bp1,
                                              ga, bta, ma, va, ba1,
                                              pscale, pshift, ascale, ashift);

  // 2) KNN indices
  knn_kernel<<<(int)(BN / 256), 256, 0, stream>>>(pos, nidx);

  // 3) xs = w_start * x + b_start   (bf16 point-major output)
  {
    int blocks = ((DIM / 16) / 8) * ((int)BN / 16);
    gemm_bf16_wmma<<<blocks, 256, 0, stream>>>(
        w_start_b, xb, b_start, DIM, INCH, (int)BN, N_,
        nullptr, xs_b, nullptr, nullptr);
  }
  // 4) k/q/v = W * xs + b          (f32 point-major outputs)
  {
    int blocks = ((DIM / 16) / 8) * ((int)BN / 16);
    gemm_bf16_wmma<<<blocks, 256, 0, stream>>>(
        w_key_b, xs_b, b_key, DIM, DIM, (int)BN, N_, kfv, nullptr, nullptr, nullptr);
    gemm_bf16_wmma<<<blocks, 256, 0, stream>>>(
        w_query_b, xs_b, b_query, DIM, DIM, (int)BN, N_, qf, nullptr, nullptr, nullptr);
    gemm_bf16_wmma<<<blocks, 256, 0, stream>>>(
        w_value_b, xs_b, b_value, DIM, DIM, (int)BN, N_, vfv, nullptr, nullptr, nullptr);
  }

  // 5) Fused gather + pos-MLP + attn-MLP + softmax + aggregate
  fused_attn_kernel<<<(int)(BN / 2), 256, 0, stream>>>(
      pos, nidx, qf, kfv, vfv, wp1, wp2, bp2, pscale, pshift,
      wa1_b, wa2_b, ascale, ashift, ba2, aggB);

  // 6) y = w_end * agg + b_end + x  (f32 channel-major into d_out)
  {
    int blocks = ((INCH / 16) / 8) * ((int)BN / 16);
    gemm_bf16_wmma<<<blocks, 256, 0, stream>>>(
        w_end_b, aggB, b_end, INCH, DIM, (int)BN, N_,
        nullptr, nullptr, out, x);
  }
}
